// HybridEncoder_75771813036527
// MI455X (gfx1250) — matrix operand order repaired and checked by reading, not compiled, not run on hardware
//
#include <hip/hip_runtime.h>
#include <hip/hip_bf16.h>
#include <math.h>

// ---------------------------------------------------------------------------
// HybridEncoder (2x [GAT -> BN+ReLU -> GCN -> BN+ReLU -> residual]) for
// MI455X / gfx1250. Dense GEMMs: TDM tensor_load_to_lds (async, TENSORcnt)
// double-buffered into LDS + V_WMMA_F32_16X16X32_F16. Sparse aggregation via
// per-edge atomics, BN via per-column LDS reductions.
// ---------------------------------------------------------------------------

typedef __attribute__((ext_vector_type(16))) _Float16     v16h;
typedef __attribute__((ext_vector_type(8)))  float        v8f;
typedef __attribute__((ext_vector_type(4)))  unsigned int uint32x4;
typedef __attribute__((ext_vector_type(8)))  int          int32x8;
typedef __attribute__((ext_vector_type(4)))  int          int32x4;

#define NNODES   20000
#define NGRAPH   64
#define NHEAD    2
#define CCH      128      // hidden channels
#define HCCH     256      // NHEAD*CCH
#define EPSBN    1e-5f
#define NEGSLOPE 0.2f
#define INVSQRT2 0.70710678118654752440f

// ---------------------------------------------------------------------------
// TDM 2-D tile load: Global (f32, row stride in elements) -> LDS, packed.
// D# layout per CDNA5 ISA (groups 0/1; 2D so groups
// 2/3 zero). data_size=4B, no multicast/iterate/pad/gather. One issue per
// wave executing this (EXEC ignored by TDM) -> call from wave 0 only.
// This toolchain uses the 6-arg builtin form (extra int32x8 group + cpol).
// ---------------------------------------------------------------------------
__device__ __forceinline__ void tdm_load_2d(unsigned lds_off, const void* gptr,
                                            unsigned tile_w, unsigned tile_h,
                                            unsigned long long row_stride_elems) {
    unsigned long long ga = (unsigned long long)(uintptr_t)gptr;
    uint32x4 g0;
    g0[0] = 1u;                                                // count=1 valid user D#
    g0[1] = lds_off;                                           // lds_addr (bytes)
    g0[2] = (unsigned)ga;                                      // global_addr[31:0]
    g0[3] = (unsigned)((ga >> 32) & 0x01FFFFFFu) | (2u << 30); // addr[56:32] | type=2
    const unsigned td0 = 0x7FFFFFFFu;                          // tensor_dim0 (generous)
    const unsigned td1 = 0x7FFFFFFFu;                          // tensor_dim1 (generous)
    unsigned long long st = row_stride_elems;                  // tensor_dim0_stride
    int32x8 g1;
    g1[0] = (int)(2u << 16);                                   // data_size=2 -> 4 bytes
    g1[1] = (int)((td0 & 0xFFFFu) << 16);                      // td0[15:0]
    g1[2] = (int)((td0 >> 16) | ((td1 & 0xFFFFu) << 16));      // td0[31:16] | td1[15:0]
    g1[3] = (int)((td1 >> 16) | (tile_w << 16));               // td1[31:16] | tile_dim0
    g1[4] = (int)(tile_h & 0xFFFFu);                           // tile_dim1 (tile_dim2=0)
    g1[5] = (int)(unsigned)st;                                 // dim0_stride[31:0]
    g1[6] = (int)((unsigned)(st >> 32) & 0xFFFFu);             // dim0_stride[47:32]
    g1[7] = 0;
    int32x4 gz4 = {0, 0, 0, 0};
    int32x8 gz8 = {0, 0, 0, 0, 0, 0, 0, 0};
    __builtin_amdgcn_tensor_load_to_lds(g0, g1, gz4, gz4, gz8, 0);
}

__device__ __forceinline__ unsigned lds_offset_of(const void* p) {
    // HIP generic pointers to __shared__ carry the LDS byte offset in addr[31:0]
    return (unsigned)(uintptr_t)p;
}

// ---------------------------------------------------------------------------
// WMMA GEMM: C[M,Nc] = A[M,K] @ B[K,Nc] (+bias) (+relu). f32 in/out, f16 MACs.
// Block = 256 threads (8 waves) -> one 16-row x 128-col output tile.
// Grid = (M/16, Nc/128). M%16==0, K%32==0, Nc%128==0 (true for all calls).
// Per 32-wide K-step, wave 0 DMAs A(16x32) and B(32x128) f32 tiles into LDS
// via TDM (double buffered), waves assemble f16 operands from LDS.
// WMMA VGPR layouts per CDNA5 ISA §7.12.2 (wave32):
//   A: m=lane%16, k = (v/4)*16 + (lane/16)*8 + (v%4)*2 + s
//   B: n=lane%16, k = (v/4)*16 + (lane/16)*8 + (v%4)*2 + s
//   D: row = v + 8*(lane/16), col = lane%16
// ---------------------------------------------------------------------------
__global__ __launch_bounds__(256)
void gemm_wmma_f16(const float* __restrict__ A, const float* __restrict__ B,
                   const float* __restrict__ bias, float* __restrict__ C,
                   int M, int K, int Nc, int relu) {
    __shared__ float Ash[2][16 * 32];
    __shared__ float Bsh[2][32 * 128];

    const int lane    = threadIdx.x & 31;
    const int wave    = threadIdx.x >> 5;
    const int half    = lane >> 4;
    const int mr      = lane & 15;
    const int row0    = blockIdx.x * 16;
    const int colBase = blockIdx.y * 128;
    const int col     = wave * 16 + mr;       // within 128-wide panel
    const bool issuer = (threadIdx.x < 32);   // wave 0 drives the TDM

    if (issuer) {
        tdm_load_2d(lds_offset_of(&Ash[0][0]), A + (size_t)row0 * K, 32, 16, (unsigned)K);
        tdm_load_2d(lds_offset_of(&Bsh[0][0]), B + colBase, 128, 32, (unsigned)Nc);
    }

    v8f acc = {};
    int buf = 0;
    for (int k0 = 0; k0 < K; k0 += 32) {
        if (issuer) __builtin_amdgcn_s_wait_tensorcnt(0);
        __syncthreads();                       // staged tiles visible to all waves

        if (issuer && (k0 + 32 < K)) {         // prefetch next K-step
            tdm_load_2d(lds_offset_of(&Ash[buf ^ 1][0]),
                        A + (size_t)row0 * K + (k0 + 32), 32, 16, (unsigned)K);
            tdm_load_2d(lds_offset_of(&Bsh[buf ^ 1][0]),
                        B + (size_t)(k0 + 32) * Nc + colBase, 128, 32, (unsigned)Nc);
        }

        const float* __restrict__ As = &Ash[buf][mr * 32];
        const float* __restrict__ Bs = &Bsh[buf][0];
        v16h av, bv;
#pragma unroll
        for (int v = 0; v < 8; ++v) {
            const int ka = ((v >> 2) << 4) + (half << 3) + ((v & 3) << 1);
            av[2 * v]     = (_Float16)As[ka];
            av[2 * v + 1] = (_Float16)As[ka + 1];
            const int kb = ((v >> 2) << 4) + (half << 3) + ((v & 3) << 1);
            bv[2 * v]     = (_Float16)Bs[kb * 128 + col];
            bv[2 * v + 1] = (_Float16)Bs[(kb + 1) * 128 + col];
        }
        acc = __builtin_amdgcn_wmma_f32_16x16x32_f16(
            /*neg_a=*/false, av, /*neg_b=*/false, bv,
            /*c_mod=*/(short)0, acc, /*reuse_a=*/false, /*reuse_b=*/false);

        __syncthreads();                       // done reading buf before reuse
        buf ^= 1;
    }

    const float bb = bias ? bias[colBase + col] : 0.0f;
#pragma unroll
    for (int v = 0; v < 8; ++v) {
        const int r = row0 + v + (half << 3);
        float val = acc[v] + bb;
        if (relu) val = fmaxf(val, 0.0f);
        C[(size_t)r * Nc + colBase + col] = val;
    }
}

// ---------------------------------------------------------------------------
// Small utility kernels
// ---------------------------------------------------------------------------
__global__ void fill_f32(float* p, float v, int n) {
    int i = blockIdx.x * blockDim.x + threadIdx.x;
    if (i < n) p[i] = v;
}

__global__ void bias_bcast(float* p, const float* __restrict__ bias, int rows, int cols) {
    int i = blockIdx.x * blockDim.x + threadIdx.x;
    if (i < rows * cols) p[i] = bias[i % cols];
}

__device__ __forceinline__ void atomicMaxF(float* addr, float val) {
    int* ia  = (int*)addr;
    int  old = __float_as_int(*addr);
    while (__int_as_float(old) < val) {
        int assumed = old;
        old = atomicCAS(ia, assumed, __float_as_int(val));
        if (old == assumed) break;
    }
}

// ---------------------------------------------------------------------------
// Degree / symmetric-norm pieces (self-loops appended logically: e>=E -> e-E)
// ---------------------------------------------------------------------------
__global__ void deg_accum(const int* __restrict__ dst, float* deg, int Etot, int E) {
    int e = blockIdx.x * blockDim.x + threadIdx.x;
    if (e >= Etot) return;
    int d = (e < E) ? dst[e] : (e - E);
    atomicAdd(&deg[d], 1.0f);
}

__global__ void deg_to_dinv(float* deg, int n) {
    int i = blockIdx.x * blockDim.x + threadIdx.x;
    if (i >= n) return;
    float d = deg[i];
    deg[i] = (d > 0.0f) ? rsqrtf(d) : 0.0f;
}

// ---------------------------------------------------------------------------
// GAT: per-node attention scores e_s, e_d  (hlin is [N, H*C] row-major)
// ---------------------------------------------------------------------------
__global__ void gat_scores(const float* __restrict__ hlin,
                           const float* __restrict__ asrc,
                           const float* __restrict__ adst,
                           float* __restrict__ es, float* __restrict__ ed) {
    int idx = blockIdx.x * blockDim.x + threadIdx.x;   // n*NHEAD + hh
    if (idx >= NNODES * NHEAD) return;
    int n = idx >> 1, hh = idx & 1;
    const float* hp = hlin + (size_t)n * HCCH + hh * CCH;
    const float* as = asrc + hh * CCH;
    const float* ad = adst + hh * CCH;
    float s = 0.0f, d = 0.0f;
    for (int c = 0; c < CCH; ++c) {
        float v = hp[c];
        s += v * as[c];
        d += v * ad[c];
    }
    es[idx] = s;
    ed[idx] = d;
}

// pass1: leaky-relu edge logits + segment max into m[dst]
__global__ void gat_edge_pass1(const int* __restrict__ src, const int* __restrict__ dst,
                               const float* __restrict__ es, const float* __restrict__ ed,
                               float* __restrict__ ee, float* m, int Etot, int E) {
    int e = blockIdx.x * blockDim.x + threadIdx.x;
    if (e >= Etot) return;
    int s = (e < E) ? src[e] : (e - E);
    int d = (e < E) ? dst[e] : (e - E);
#pragma unroll
    for (int hh = 0; hh < NHEAD; ++hh) {
        float v = es[s * NHEAD + hh] + ed[d * NHEAD + hh];
        v = (v > 0.0f) ? v : NEGSLOPE * v;
        ee[(size_t)e * NHEAD + hh] = v;
        atomicMaxF(&m[d * NHEAD + hh], v);
    }
}

// pass2: ex = exp(e - m[dst]); den[dst] += ex  (ee overwritten in place)
__global__ void gat_edge_pass2(const int* __restrict__ dst,
                               float* __restrict__ ee, const float* __restrict__ m,
                               float* den, int Etot, int E) {
    int e = blockIdx.x * blockDim.x + threadIdx.x;
    if (e >= Etot) return;
    int d = (e < E) ? dst[e] : (e - E);
#pragma unroll
    for (int hh = 0; hh < NHEAD; ++hh) {
        float ex = __expf(ee[(size_t)e * NHEAD + hh] - m[d * NHEAD + hh]);
        ee[(size_t)e * NHEAD + hh] = ex;
        atomicAdd(&den[d * NHEAD + hh], ex);
    }
}

// pass3: out[dst] += h[src] * alpha   (one block per edge, 256 channels)
__global__ __launch_bounds__(HCCH)
void gat_edge_pass3(const int* __restrict__ src, const int* __restrict__ dst,
                    const float* __restrict__ hlin, const float* __restrict__ ee,
                    const float* __restrict__ den, float* outp, int Etot, int E) {
    int e = blockIdx.x;
    if (e >= Etot) return;
    int t  = threadIdx.x;            // hh*128 + c
    int s  = (e < E) ? src[e] : (e - E);
    int d  = (e < E) ? dst[e] : (e - E);
    int hh = t >> 7;
    float alpha = ee[(size_t)e * NHEAD + hh] / (den[d * NHEAD + hh] + 1e-16f);
    atomicAdd(&outp[(size_t)d * HCCH + t], hlin[(size_t)s * HCCH + t] * alpha);
}

// GCN scatter: out[dst] += g[src] * dinv[src]*dinv[dst]  (128 channels/edge)
__global__ __launch_bounds__(CCH)
void gcn_scatter(const int* __restrict__ src, const int* __restrict__ dst,
                 const float* __restrict__ g, const float* __restrict__ dinv,
                 float* outp, int Etot, int E) {
    int e = blockIdx.x;
    if (e >= Etot) return;
    int t = threadIdx.x;
    int s = (e < E) ? src[e] : (e - E);
    int d = (e < E) ? dst[e] : (e - E);
    float nrm = dinv[s] * dinv[d];
    atomicAdd(&outp[(size_t)d * CCH + t], g[(size_t)s * CCH + t] * nrm);
}

// ---------------------------------------------------------------------------
// BatchNorm (training-mode batch stats): per-column mean + rsqrt(var+eps)
// ---------------------------------------------------------------------------
__global__ __launch_bounds__(256)
void bn_stats(const float* __restrict__ X, int rows, int cols, float* mu, float* rs) {
    __shared__ float ssum[256];
    __shared__ float ssq[256];
    int j = blockIdx.x;                        // column
    float s = 0.0f, q = 0.0f;
    for (int i = threadIdx.x; i < rows; i += 256) {
        float v = X[(size_t)i * cols + j];
        s += v; q += v * v;
    }
    ssum[threadIdx.x] = s; ssq[threadIdx.x] = q;
    __syncthreads();
    for (int st = 128; st > 0; st >>= 1) {
        if (threadIdx.x < st) {
            ssum[threadIdx.x] += ssum[threadIdx.x + st];
            ssq[threadIdx.x]  += ssq[threadIdx.x + st];
        }
        __syncthreads();
    }
    if (threadIdx.x == 0) {
        float mean = ssum[0] / (float)rows;
        float var  = ssq[0] / (float)rows - mean * mean;
        mu[j] = mean;
        rs[j] = rsqrtf(var + EPSBN);
    }
}

__global__ void bn_apply_relu(const float* __restrict__ X,
                              const float* __restrict__ mu, const float* __restrict__ rs,
                              const float* __restrict__ g,  const float* __restrict__ be,
                              float* Y, int rows, int cols) {
    int i = blockIdx.x * blockDim.x + threadIdx.x;
    if (i >= rows * cols) return;
    int j = i % cols;
    float v = (X[i] - mu[j]) * rs[j] * g[j] + be[j];
    Y[i] = fmaxf(v, 0.0f);
}

// z = relu(bn(x)); h = (z + h) * 1/sqrt(2)   (residual update in place)
__global__ void bn_apply_relu_res(const float* __restrict__ X,
                                  const float* __restrict__ mu, const float* __restrict__ rs,
                                  const float* __restrict__ g,  const float* __restrict__ be,
                                  float* H, int rows, int cols) {
    int i = blockIdx.x * blockDim.x + threadIdx.x;
    if (i >= rows * cols) return;
    int j = i % cols;
    float v = (X[i] - mu[j]) * rs[j] * g[j] + be[j];
    v = fmaxf(v, 0.0f);
    H[i] = (v + H[i]) * INVSQRT2;
}

// ---------------------------------------------------------------------------
// Global mean pool
// ---------------------------------------------------------------------------
__global__ void pool_accum(const float* __restrict__ H, const int* __restrict__ batch,
                           float* sums) {
    int i = blockIdx.x * blockDim.x + threadIdx.x;
    if (i >= NNODES * CCH) return;
    int n = i >> 7, c = i & 127;
    atomicAdd(&sums[batch[n] * CCH + c], H[i]);
}

__global__ void pool_count(const int* __restrict__ batch, float* cnt) {
    int n = blockIdx.x * blockDim.x + threadIdx.x;
    if (n < NNODES) atomicAdd(&cnt[batch[n]], 1.0f);
}

__global__ void pool_divide(float* sums, const float* __restrict__ cnt) {
    int i = blockIdx.x * blockDim.x + threadIdx.x;
    if (i >= NGRAPH * CCH) return;
    sums[i] /= fmaxf(cnt[i >> 7], 1.0f);
}

// ---------------------------------------------------------------------------
// Host orchestration
// ---------------------------------------------------------------------------
extern "C" void kernel_launch(void* const* d_in, const int* in_sizes, int n_in,
                              void* d_out, int out_size, void* d_ws, size_t ws_size,
                              hipStream_t stream) {
    const float* x    = (const float*)d_in[0];
    const int*   ei   = (const int*)d_in[1];      // [2,E]
    const int*   bat  = (const int*)d_in[2];
    const float* W1   = (const float*)d_in[3];    // [NB,128,256]
    const float* asrc = (const float*)d_in[4];    // [NB,2,128]
    const float* adst = (const float*)d_in[5];
    const float* b1   = (const float*)d_in[6];    // [NB,256]
    const float* g1   = (const float*)d_in[7];
    const float* be1  = (const float*)d_in[8];
    const float* W2   = (const float*)d_in[9];    // [NB,256,128]
    const float* b2   = (const float*)d_in[10];   // [NB,128]
    const float* g2   = (const float*)d_in[11];
    const float* be2  = (const float*)d_in[12];
    const float* Wh1  = (const float*)d_in[13];   // [128,256]
    const float* bh1  = (const float*)d_in[14];
    const float* Wh2  = (const float*)d_in[15];   // [256,128]
    const float* bh2  = (const float*)d_in[16];

    const int E    = in_sizes[1] / 2;
    const int Etot = E + NNODES;
    const int* src = ei;
    const int* dst = ei + E;

    // workspace layout (floats)
    float* W = (float*)d_ws;
    size_t o = 0;
    float* h_lin = W + o; o += (size_t)NNODES * HCCH;   // GAT lin out / GCN lin out
    float* attn  = W + o; o += (size_t)NNODES * HCCH;   // GAT aggregation out / bn1 out
    float* z2    = W + o; o += (size_t)NNODES * CCH;    // GCN aggregation out
    float* hcur  = W + o; o += (size_t)NNODES * CCH;    // running node features
    float* dinv  = W + o; o += NNODES;                  // deg -> d^-1/2
    float* es    = W + o; o += (size_t)NNODES * NHEAD;
    float* ed    = W + o; o += (size_t)NNODES * NHEAD;
    float* mm    = W + o; o += (size_t)NNODES * NHEAD;
    float* den   = W + o; o += (size_t)NNODES * NHEAD;
    float* ee    = W + o; o += (size_t)Etot * NHEAD;
    float* mu    = W + o; o += HCCH;
    float* rs    = W + o; o += HCCH;
    float* sums  = W + o; o += NGRAPH * CCH;
    float* cnt   = W + o; o += NGRAPH;
    float* hid   = W + o; o += NGRAPH * 256;

    const int TB = 256;
    auto cdiv = [](int a, int b) { return (a + b - 1) / b; };

    // h = x (don't mutate inputs)
    (void)hipMemcpyAsync(hcur, x, (size_t)NNODES * CCH * sizeof(float),
                         hipMemcpyDeviceToDevice, stream);

    // degrees (with self-loops) -> d^-1/2
    fill_f32<<<cdiv(NNODES, TB), TB, 0, stream>>>(dinv, 0.0f, NNODES);
    deg_accum<<<cdiv(Etot, TB), TB, 0, stream>>>(dst, dinv, Etot, E);
    deg_to_dinv<<<cdiv(NNODES, TB), TB, 0, stream>>>(dinv, NNODES);

    for (int b = 0; b < 2; ++b) {
        const float* W1b = W1 + (size_t)b * CCH * HCCH;
        const float* W2b = W2 + (size_t)b * HCCH * CCH;

        // ---- GAT ----
        // h_lin = hcur @ W1b   [N,128]x[128,256]
        gemm_wmma_f16<<<dim3(NNODES / 16, HCCH / 128), 256, 0, stream>>>(
            hcur, W1b, nullptr, h_lin, NNODES, CCH, HCCH, 0);
        gat_scores<<<cdiv(NNODES * NHEAD, TB), TB, 0, stream>>>(
            h_lin, asrc + (size_t)b * NHEAD * CCH, adst + (size_t)b * NHEAD * CCH, es, ed);
        fill_f32<<<cdiv(NNODES * NHEAD, TB), TB, 0, stream>>>(mm, -3.0e38f, NNODES * NHEAD);
        fill_f32<<<cdiv(NNODES * NHEAD, TB), TB, 0, stream>>>(den, 0.0f, NNODES * NHEAD);
        bias_bcast<<<cdiv(NNODES * HCCH, TB), TB, 0, stream>>>(
            attn, b1 + (size_t)b * HCCH, NNODES, HCCH);
        gat_edge_pass1<<<cdiv(Etot, TB), TB, 0, stream>>>(src, dst, es, ed, ee, mm, Etot, E);
        gat_edge_pass2<<<cdiv(Etot, TB), TB, 0, stream>>>(dst, ee, mm, den, Etot, E);
        gat_edge_pass3<<<Etot, HCCH, 0, stream>>>(src, dst, h_lin, ee, den, attn, Etot, E);

        // bn1 + relu (in place on attn)
        bn_stats<<<HCCH, 256, 0, stream>>>(attn, NNODES, HCCH, mu, rs);
        bn_apply_relu<<<cdiv(NNODES * HCCH, TB), TB, 0, stream>>>(
            attn, mu, rs, g1 + (size_t)b * HCCH, be1 + (size_t)b * HCCH,
            attn, NNODES, HCCH);

        // ---- GCN ----
        // h_lin(reused, first N*128) = attn @ W2b   [N,256]x[256,128]
        gemm_wmma_f16<<<dim3(NNODES / 16, CCH / 128), 256, 0, stream>>>(
            attn, W2b, nullptr, h_lin, NNODES, HCCH, CCH, 0);
        bias_bcast<<<cdiv(NNODES * CCH, TB), TB, 0, stream>>>(
            z2, b2 + (size_t)b * CCH, NNODES, CCH);
        gcn_scatter<<<Etot, CCH, 0, stream>>>(src, dst, h_lin, dinv, z2, Etot, E);

        // bn2 + relu + residual into hcur
        bn_stats<<<CCH, 256, 0, stream>>>(z2, NNODES, CCH, mu, rs);
        bn_apply_relu_res<<<cdiv(NNODES * CCH, TB), TB, 0, stream>>>(
            z2, mu, rs, g2 + (size_t)b * CCH, be2 + (size_t)b * CCH,
            hcur, NNODES, CCH);
    }

    // global mean pool
    fill_f32<<<cdiv(NGRAPH * CCH, TB), TB, 0, stream>>>(sums, 0.0f, NGRAPH * CCH);
    fill_f32<<<cdiv(NGRAPH, TB), TB, 0, stream>>>(cnt, 0.0f, NGRAPH);
    pool_accum<<<cdiv(NNODES * CCH, TB), TB, 0, stream>>>(hcur, bat, sums);
    pool_count<<<cdiv(NNODES, TB), TB, 0, stream>>>(bat, cnt);
    pool_divide<<<cdiv(NGRAPH * CCH, TB), TB, 0, stream>>>(sums, cnt);

    // MLP head: relu(pooled @ Wh1 + bh1) @ Wh2 + bh2
    gemm_wmma_f16<<<dim3(NGRAPH / 16, 256 / 128), 256, 0, stream>>>(
        sums, Wh1, bh1, hid, NGRAPH, CCH, 256, 1);
    gemm_wmma_f16<<<dim3(NGRAPH / 16, CCH / 128), 256, 0, stream>>>(
        hid, Wh2, bh2, (float*)d_out, NGRAPH, 256, CCH, 0);
}
